// MegaCRN_41944650613023
// MI455X (gfx1250) — matrix operand order repaired and checked by reading, not compiled, not run on hardware
//
#include <hip/hip_runtime.h>
#include <hip/hip_bf16.h>
#include <math.h>

// ---------------- model constants (from reference) ----------------
constexpr int N_    = 1024;
constexpr int B_    = 32;
constexpr int T_    = 12;
constexpr int HOR_  = 12;
constexpr int HID_  = 64;
constexpr int EMB_  = 8;
constexpr int KCH_  = 3;     // Chebyshev order
constexpr int MEMN_ = 10;
constexpr int MEMD_ = 32;
constexpr int DECD_ = 96;    // HID + MEM_D

typedef __attribute__((ext_vector_type(16))) __bf16 v16bf;
typedef __attribute__((ext_vector_type(8)))  float  v8f;

// ---------------- epilogue helpers ----------------
__device__ __forceinline__ float gemm_epilogue(float acc, int epi, int row, int col,
                                               const float* __restrict__ bias) {
  if (epi == 1) return 2.0f * acc - ((row == col) ? 1.0f : 0.0f);   // Chebyshev T2
  if (epi == 2) { float x = acc + bias[col]; return 1.0f / (1.0f + __expf(-x)); }
  if (epi == 3) { float x = acc + bias[col]; return tanhf(x); }
  return acc;
}

__device__ __forceinline__ void store_c(void* C, int cT, size_t off, float val) {
  if (cT) ((__bf16*)C)[off] = (__bf16)val;
  else    ((float*)C)[off]  = val;
}

// ---------------- generic bf16-WMMA GEMM: C = A @ B (+epilogue) ----------------
// Block tile 128x64, 8 waves each computing a 32x32 tile (2x2 WMMA subtiles,
// 4 v_wmma per K-step, A/B fragments each reused twice).
// aT/bT/cT: 0 = fp32 in memory, 1 = bf16 in memory.
// lhs_mode 0: A row-major (lda). lhs_mode 1: AGCN gather — logical row r=(b*N+n),
// logical col q=(k*c+j), element = base[((k*g_N+n)*g_Bc) + b*g_c + j].
#define GEMM_BM 128
#define GEMM_BN 64
#define GEMM_BK 32

__global__ void __launch_bounds__(256)
wmma_gemm_kernel(const void* __restrict__ Aop, const void* __restrict__ Bop,
                 void* __restrict__ Cop, const float* __restrict__ bias,
                 int M, int Nn, int Kd, int lda, int ldb, int ldc,
                 int epi, int aT, int bT, int cT,
                 int lhs_mode, int g_c, int g_Bc, int g_N)
{
  __shared__ alignas(16) __bf16 sA[GEMM_BM * GEMM_BK];
  __shared__ alignas(16) __bf16 sB[GEMM_BK * GEMM_BN];

  const float*  Af = (const float*)Aop;
  const __bf16* Ab = (const __bf16*)Aop;
  const float*  Bf = (const float*)Bop;
  const __bf16* Bb = (const __bf16*)Bop;

  const int tid  = threadIdx.x;
  const int lane = tid & 31;
  const int wave = tid >> 5;            // 0..7
  const int bm0  = blockIdx.y * GEMM_BM;
  const int bn0  = blockIdx.x * GEMM_BN;
  const int mW   = wave & 3;            // 4 row groups of 32
  const int nW   = wave >> 2;           // 2 col groups of 32

  v8f acc00 = {}, acc01 = {}, acc10 = {}, acc11 = {};

  for (int kb = 0; kb < Kd; kb += GEMM_BK) {
    // ---- stage A tile ----
    if (aT && lhs_mode == 0) {
      // raw bf16 16-byte copies (b128 load -> b128 LDS store)
      for (int i = tid; i < (GEMM_BM * GEMM_BK) / 8; i += 256) {
        int row = i >> 2, seg = (i & 3) * 8;
        int gr = bm0 + row, gk = kb + seg;
        uint4 v4 = make_uint4(0u, 0u, 0u, 0u);
        if (gr < M) {
          if (gk + 8 <= Kd) {
            v4 = *(const uint4*)(Ab + (size_t)gr * lda + gk);
          } else {
            alignas(16) __bf16 tmp[8] = {};
            for (int e = 0; e < 8; ++e)
              if (gk + e < Kd) tmp[e] = Ab[(size_t)gr * lda + gk + e];
            v4 = *(const uint4*)tmp;
          }
        }
        *(uint4*)&sA[row * GEMM_BK + seg] = v4;
      }
    } else {
      for (int i = tid; i < GEMM_BM * GEMM_BK; i += 256) {
        int row = i >> 5, col = i & 31;
        int gr = bm0 + row, gk = kb + col;
        __bf16 v = (__bf16)0.0f;
        if (gr < M && gk < Kd) {
          size_t idx;
          if (lhs_mode == 0) {
            idx = (size_t)gr * lda + gk;
          } else {
            int b = gr / g_N, n = gr - b * g_N;
            int k = gk / g_c, j = gk - k * g_c;
            idx = ((size_t)k * g_N + n) * g_Bc + (size_t)b * g_c + j;
          }
          v = aT ? Ab[idx] : (__bf16)Af[idx];
        }
        sA[i] = v;
      }
    }
    // ---- stage B tile ----
    if (bT) {
      for (int i = tid; i < (GEMM_BK * GEMM_BN) / 8; i += 256) {  // exactly 1 iter
        int row = i >> 3, seg = (i & 7) * 8;
        int gk = kb + row, gc = bn0 + seg;
        uint4 v4 = make_uint4(0u, 0u, 0u, 0u);
        if (gk < Kd) {
          if (gc + 8 <= Nn) {
            v4 = *(const uint4*)(Bb + (size_t)gk * ldb + gc);
          } else {
            alignas(16) __bf16 tmp[8] = {};
            for (int e = 0; e < 8; ++e)
              if (gc + e < Nn) tmp[e] = Bb[(size_t)gk * ldb + gc + e];
            v4 = *(const uint4*)tmp;
          }
        }
        *(uint4*)&sB[row * GEMM_BN + seg] = v4;
      }
    } else {
      for (int i = tid; i < GEMM_BK * GEMM_BN; i += 256) {
        int row = i >> 6, col = i & 63;
        int gk = kb + row, gc = bn0 + col;
        float v = 0.0f;
        if (gk < Kd && gc < Nn) v = Bf[(size_t)gk * ldb + gc];
        sB[i] = (__bf16)v;
      }
    }
    // prefetch next K-tile of B (one cacheline per wave) -> global_prefetch_b8
    if (lane == 0) {
      int prow = kb + GEMM_BK + wave * 4;
      if (prow < Kd) {
        const void* pf = bT ? (const void*)(Bb + (size_t)prow * ldb + bn0)
                            : (const void*)(Bf + (size_t)prow * ldb + bn0);
        __builtin_prefetch(pf, 0, 1);
      }
    }
    __syncthreads();

    // ---- build fragments per CDNA5 ISA layouts ----
    v16bf a0, a1, b0v, b1v;
    {
      int mrow0 = mW * 32 + (lane & 15);
      int mrow1 = mrow0 + 16;
      int kbase = (lane < 16) ? 0 : 8;
      int n0 = nW * 32 + (lane & 15);
#pragma unroll
      for (int i = 0; i < 8; ++i) {
        a0[i]     = sA[mrow0 * GEMM_BK + kbase + i];
        a0[8 + i] = sA[mrow0 * GEMM_BK + 16 + kbase + i];
        a1[i]     = sA[mrow1 * GEMM_BK + kbase + i];
        a1[8 + i] = sA[mrow1 * GEMM_BK + 16 + kbase + i];
        b0v[i]     = sB[(kbase + i) * GEMM_BN + n0];
        b0v[8 + i] = sB[(16 + kbase + i) * GEMM_BN + n0];
        b1v[i]     = sB[(kbase + i) * GEMM_BN + n0 + 16];
        b1v[8 + i] = sB[(16 + kbase + i) * GEMM_BN + n0 + 16];
      }
    }
    acc00 = __builtin_amdgcn_wmma_f32_16x16x32_bf16(false, a0, false, b0v, (short)0, acc00, false, false);
    acc01 = __builtin_amdgcn_wmma_f32_16x16x32_bf16(false, a0, false, b1v, (short)0, acc01, false, false);
    acc10 = __builtin_amdgcn_wmma_f32_16x16x32_bf16(false, a1, false, b0v, (short)0, acc10, false, false);
    acc11 = __builtin_amdgcn_wmma_f32_16x16x32_bf16(false, a1, false, b1v, (short)0, acc11, false, false);
    __syncthreads();
  }

  // ---- epilogue / writeback (C/D layout: VGPR v -> M = v + 8*(lane>=16), N = lane%16) ----
  const int moff = (lane < 16) ? 0 : 8;
  const int ncA  = bn0 + nW * 32 + (lane & 15);
  const int ncB  = ncA + 16;
  const int rb0  = bm0 + mW * 32;
  const int rb1  = rb0 + 16;
#pragma unroll
  for (int vv = 0; vv < 8; ++vv) {
    int r0 = rb0 + moff + vv;
    int r1 = rb1 + moff + vv;
    if (r0 < M) {
      if (ncA < Nn) store_c(Cop, cT, (size_t)r0 * ldc + ncA, gemm_epilogue(acc00[vv], epi, r0, ncA, bias));
      if (ncB < Nn) store_c(Cop, cT, (size_t)r0 * ldc + ncB, gemm_epilogue(acc01[vv], epi, r0, ncB, bias));
    }
    if (r1 < M) {
      if (ncA < Nn) store_c(Cop, cT, (size_t)r1 * ldc + ncA, gemm_epilogue(acc10[vv], epi, r1, ncA, bias));
      if (ncB < Nn) store_c(Cop, cT, (size_t)r1 * ldc + ncB, gemm_epilogue(acc11[vv], epi, r1, ncB, bias));
    }
  }
}

// ---------------- A = softmax_row(relu(E @ E^T)) -> bf16, one block per row ----------------
__global__ void __launch_bounds__(256)
gram_softmax_kernel(const float* __restrict__ E, int D, int ldE,
                    __bf16* __restrict__ Aout, int Nn)
{
  __shared__ float se[256];
  __shared__ float red[256];
  const int n = blockIdx.x;
  const int tid = threadIdx.x;

  for (int i = tid; i < D; i += 256) se[i] = E[(size_t)n * ldE + i];
  __syncthreads();

  float vals[4];
  float vmax = -1e30f;
#pragma unroll
  for (int u = 0; u < 4; ++u) {
    int m = tid + u * 256;
    float dot = 0.0f;
    for (int d = 0; d < D; ++d) dot += se[d] * E[(size_t)m * ldE + d];
    float r = dot > 0.0f ? dot : 0.0f;   // relu
    vals[u] = r;
    vmax = fmaxf(vmax, r);
  }
  red[tid] = vmax; __syncthreads();
  for (int s = 128; s > 0; s >>= 1) { if (tid < s) red[tid] = fmaxf(red[tid], red[tid + s]); __syncthreads(); }
  vmax = red[0]; __syncthreads();

  float vsum = 0.0f;
#pragma unroll
  for (int u = 0; u < 4; ++u) { vals[u] = __expf(vals[u] - vmax); vsum += vals[u]; }
  red[tid] = vsum; __syncthreads();
  for (int s = 128; s > 0; s >>= 1) { if (tid < s) red[tid] += red[tid + s]; __syncthreads(); }
  float inv = 1.0f / red[0];

#pragma unroll
  for (int u = 0; u < 4; ++u)
    Aout[(size_t)n * Nn + tid + u * 256] = (__bf16)(vals[u] * inv);
}

// ---------------- fp32 -> bf16 converter (weights, once per launch) ----------------
__global__ void f32_to_bf16_kernel(const float* __restrict__ src, __bf16* __restrict__ dst, int n)
{
  int i = blockIdx.x * blockDim.x + threadIdx.x;
  if (i < n) dst[i] = (__bf16)src[i];
}

// ---------------- pack [x|h] (or [go|ycov|h], optionally z*h) into Xg slot 0 (bf16) ----------------
__global__ void pack_kernel(__bf16* __restrict__ Xg0, const float* __restrict__ hist, int t,
                            int is_dec, const float* __restrict__ go,
                            const float* __restrict__ h, const float* __restrict__ z,
                            int zr_stride, int hid, int c, int Bc, int total)
{
  int idx = blockIdx.x * blockDim.x + threadIdx.x;
  if (idx >= total) return;
  int m = idx / Bc;
  int rem = idx - m * Bc;
  int b = rem / c;
  int j = rem - b * c;
  int bn = b * N_ + m;
  float v;
  if (is_dec) {
    if (j == 0)      v = go[bn];
    else if (j == 1) v = hist[(((size_t)b * T_ + t) * N_ + m) * 2 + 1];
    else {
      int jj = j - 2;
      v = h[(size_t)bn * hid + jj];
      if (z) v *= z[(size_t)bn * zr_stride + jj];
    }
  } else {
    if (j == 0) v = hist[(((size_t)b * T_ + t) * N_ + m) * 2 + 0];
    else {
      int jj = j - 1;
      v = h[(size_t)bn * hid + jj];
      if (z) v *= z[(size_t)bn * zr_stride + jj];
    }
  }
  Xg0[idx] = (__bf16)v;
}

// ---------------- GRU state combine: h = r*h + (1-r)*hc ----------------
__global__ void gru_combine_kernel(float* __restrict__ h, const float* __restrict__ zr,
                                   const float* __restrict__ hc, int hid, int total)
{
  int idx = blockIdx.x * blockDim.x + threadIdx.x;
  if (idx >= total) return;
  int bn = idx / hid;
  int j  = idx - bn * hid;
  float r = zr[(size_t)bn * (2 * hid) + hid + j];
  h[idx] = r * h[idx] + (1.0f - r) * hc[idx];
}

// ---------------- fused memory attention: softmax, proto, top-2, W_E, h_cat ----------------
__global__ void __launch_bounds__(256)
attention_kernel(const float* __restrict__ h_enc, const float* __restrict__ Mem,
                 const float* __restrict__ FCE, const float* __restrict__ query,
                 float* __restrict__ proto_out, float* __restrict__ pos_out,
                 float* __restrict__ neg_out, float* __restrict__ WE2,
                 float* __restrict__ h_cat)
{
  __shared__ float sMem[MEMN_ * MEMD_];
  __shared__ float sFCE[MEMD_ * EMB_];
  int tid = threadIdx.x;
  for (int i = tid; i < MEMN_ * MEMD_; i += 256) sMem[i] = Mem[i];
  for (int i = tid; i < MEMD_ * EMB_;  i += 256) sFCE[i] = FCE[i];
  __syncthreads();

  int bn = blockIdx.x * 256 + tid;
  int b = bn / N_, n = bn - b * N_;

  float q[MEMD_];
#pragma unroll
  for (int j = 0; j < MEMD_; ++j) q[j] = query[(size_t)bn * MEMD_ + j];

  float lg[MEMN_];
  float lmax = -1e30f;
#pragma unroll
  for (int i = 0; i < MEMN_; ++i) {
    float d = 0.0f;
#pragma unroll
    for (int j = 0; j < MEMD_; ++j) d += q[j] * sMem[i * MEMD_ + j];
    lg[i] = d; lmax = fmaxf(lmax, d);
  }
  float s = 0.0f;
#pragma unroll
  for (int i = 0; i < MEMN_; ++i) { lg[i] = __expf(lg[i] - lmax); s += lg[i]; }
  float inv = 1.0f / s;

  int i0 = 0, i1 = -1;
  float a0 = -1.0f, a1 = -1.0f;
#pragma unroll
  for (int i = 0; i < MEMN_; ++i) {
    float a = lg[i] * inv; lg[i] = a;
    if (a > a0)      { a1 = a0; i1 = i0; a0 = a; i0 = i; }
    else if (a > a1) { a1 = a;  i1 = i; }
  }

  float proto[MEMD_];
#pragma unroll
  for (int j = 0; j < MEMD_; ++j) {
    float p = 0.0f;
#pragma unroll
    for (int i = 0; i < MEMN_; ++i) p += lg[i] * sMem[i * MEMD_ + j];
    proto[j] = p;
    proto_out[(size_t)bn * MEMD_ + j] = p;
    pos_out[(size_t)bn * MEMD_ + j]   = sMem[i0 * MEMD_ + j];
    neg_out[(size_t)bn * MEMD_ + j]   = sMem[i1 * MEMD_ + j];
  }
  // W_E = proto @ FC_E, stored as (N, B*EMB) for the decoder gram
#pragma unroll
  for (int c = 0; c < EMB_; ++c) {
    float w = 0.0f;
#pragma unroll
    for (int j = 0; j < MEMD_; ++j) w += proto[j] * sFCE[j * EMB_ + c];
    WE2[(size_t)n * (B_ * EMB_) + b * EMB_ + c] = w;
  }
  // h_cat = [h_t | proto]
#pragma unroll
  for (int j = 0; j < HID_; ++j) h_cat[(size_t)bn * DECD_ + j] = h_enc[(size_t)bn * HID_ + j];
#pragma unroll
  for (int j = 0; j < MEMD_; ++j) h_cat[(size_t)bn * DECD_ + HID_ + j] = proto[j];
}

// ---------------- decoder output projection ----------------
__global__ void proj_kernel(const float* __restrict__ h_cat, const float* __restrict__ pw,
                            const float* __restrict__ pb, float* __restrict__ go,
                            float* __restrict__ out, int t)
{
  int bn = blockIdx.x * blockDim.x + threadIdx.x;
  if (bn >= B_ * N_) return;
  int b = bn / N_, n = bn - b * N_;
  float acc = pb[0];
  for (int j = 0; j < DECD_; ++j) acc += h_cat[(size_t)bn * DECD_ + j] * pw[j];
  go[bn] = acc;
  out[((size_t)b * HOR_ + t) * N_ + n] = acc;
}

// ---------------- host orchestration ----------------
extern "C" void kernel_launch(void* const* d_in, const int* in_sizes, int n_in,
                              void* d_out, int out_size, void* d_ws, size_t ws_size,
                              hipStream_t stream) {
  (void)in_sizes; (void)n_in; (void)out_size; (void)ws_size;
  const float* hist = (const float*)d_in[0];
  const float* emb  = (const float*)d_in[1];
  const float* Mem  = (const float*)d_in[2];
  const float* Wq   = (const float*)d_in[3];
  const float* FCE  = (const float*)d_in[4];
  const float* egw  = (const float*)d_in[5];
  const float* egb  = (const float*)d_in[6];
  const float* euw  = (const float*)d_in[7];
  const float* eub  = (const float*)d_in[8];
  const float* dgw  = (const float*)d_in[9];
  const float* dgb  = (const float*)d_in[10];
  const float* duw  = (const float*)d_in[11];
  const float* dub  = (const float*)d_in[12];
  const float* pw   = (const float*)d_in[13];
  const float* pb   = (const float*)d_in[14];
  float* out = (float*)d_out;

  // d_out regions: (output, proto, query, pos, neg)
  const size_t OFF_OUT   = 0;
  const size_t OFF_PROTO = (size_t)B_ * HOR_ * N_;
  const size_t OFF_QUERY = OFF_PROTO + (size_t)B_ * N_ * MEMD_;
  const size_t OFF_POS   = OFF_QUERY + (size_t)B_ * N_ * MEMD_;
  const size_t OFF_NEG   = OFF_POS   + (size_t)B_ * N_ * MEMD_;

  const size_t NN = (size_t)N_ * N_;
  const size_t BN = (size_t)B_ * N_;
  const int BC_MAX = B_ * 98;

  // workspace carve-out (256B aligned regions)
  char* wp = (char*)d_ws;
  auto alloc = [&](size_t bytes) -> char* {
    char* r = wp; wp += (bytes + 255) & ~(size_t)255; return r;
  };
  __bf16* A_enc  = (__bf16*)alloc(NN * 2);
  __bf16* T2_enc = (__bf16*)alloc(NN * 2);
  __bf16* A_dec  = (__bf16*)alloc(NN * 2);
  __bf16* T2_dec = (__bf16*)alloc(NN * 2);
  __bf16* Xg     = (__bf16*)alloc((size_t)3 * N_ * BC_MAX * 2);
  __bf16* egw_b  = (__bf16*)alloc((size_t)195 * 128 * 2);
  __bf16* euw_b  = (__bf16*)alloc((size_t)195 * 64 * 2);
  __bf16* dgw_b  = (__bf16*)alloc((size_t)294 * 192 * 2);
  __bf16* duw_b  = (__bf16*)alloc((size_t)294 * 96 * 2);
  __bf16* Wq_b   = (__bf16*)alloc((size_t)HID_ * MEMD_ * 2);
  float* zr      = (float*)alloc(BN * 192 * 4);
  float* h_enc   = (float*)alloc(BN * HID_ * 4);
  float* h_cat   = (float*)alloc(BN * DECD_ * 4);
  float* hc      = (float*)alloc(BN * DECD_ * 4);
  float* WE2     = (float*)alloc((size_t)N_ * (B_ * EMB_) * 4);
  float* go      = (float*)alloc(BN * 4);

  auto gemm = [&](const void* A, const void* Bm, void* C, const float* bias,
                  int M, int Nn, int Kd, int lda, int ldb, int ldc,
                  int epi, int aT, int bT, int cT, int lhs_mode, int g_c, int g_Bc) {
    dim3 grid((Nn + GEMM_BN - 1) / GEMM_BN, (M + GEMM_BM - 1) / GEMM_BM);
    wmma_gemm_kernel<<<grid, 256, 0, stream>>>(A, Bm, C, bias, M, Nn, Kd,
                                               lda, ldb, ldc, epi, aT, bT, cT,
                                               lhs_mode, g_c, g_Bc, N_);
  };
  auto cvt = [&](const float* s, __bf16* d, int n) {
    f32_to_bf16_kernel<<<(n + 255) / 256, 256, 0, stream>>>(s, d, n);
  };

  // ----- one-time weight conversions to bf16 -----
  cvt(egw, egw_b, 195 * 128);
  cvt(euw, euw_b, 195 * 64);
  cvt(dgw, dgw_b, 294 * 192);
  cvt(duw, duw_b, 294 * 96);
  cvt(Wq,  Wq_b,  HID_ * MEMD_);

  // ----- encoder supports: A = softmax(relu(E E^T)), T2 = 2 A A - I -----
  gram_softmax_kernel<<<N_, 256, 0, stream>>>(emb, EMB_, EMB_, A_enc, N_);
  gemm(A_enc, A_enc, T2_enc, nullptr, N_, N_, N_, N_, N_, N_, 1, 1, 1, 1, 0, 0, 0);

  hipMemsetAsync(h_enc, 0, BN * HID_ * sizeof(float), stream);

  // ----- encoder scan -----
  {
    const int c = 1 + HID_, Bc = B_ * c, Kd = KCH_ * c;   // 65, 2080, 195
    __bf16* Xg0 = Xg;
    __bf16* Xg1 = Xg0 + (size_t)N_ * Bc;
    __bf16* Xg2 = Xg1 + (size_t)N_ * Bc;
    const int packTotal = N_ * Bc;
    const int packBlocks = (packTotal + 255) / 256;
    for (int t = 0; t < T_; ++t) {
      // gate path
      pack_kernel<<<packBlocks, 256, 0, stream>>>(Xg0, hist, t, 0, nullptr, h_enc, nullptr, 0, HID_, c, Bc, packTotal);
      gemm(A_enc,  Xg0, Xg1, nullptr, N_, Bc, N_, N_, Bc, Bc, 0, 1, 1, 1, 0, 0, 0);
      gemm(T2_enc, Xg0, Xg2, nullptr, N_, Bc, N_, N_, Bc, Bc, 0, 1, 1, 1, 0, 0, 0);
      gemm(Xg, egw_b, zr, egb, B_ * N_, 2 * HID_, Kd, 0, 2 * HID_, 2 * HID_, 2, 1, 1, 0, 1, c, Bc);
      // update path (x | z*h)
      pack_kernel<<<packBlocks, 256, 0, stream>>>(Xg0, hist, t, 0, nullptr, h_enc, zr, 2 * HID_, HID_, c, Bc, packTotal);
      gemm(A_enc,  Xg0, Xg1, nullptr, N_, Bc, N_, N_, Bc, Bc, 0, 1, 1, 1, 0, 0, 0);
      gemm(T2_enc, Xg0, Xg2, nullptr, N_, Bc, N_, N_, Bc, Bc, 0, 1, 1, 1, 0, 0, 0);
      gemm(Xg, euw_b, hc, eub, B_ * N_, HID_, Kd, 0, HID_, HID_, 3, 1, 1, 0, 1, c, Bc);
      // h = r*h + (1-r)*hc
      int tot = (int)(BN * HID_);
      gru_combine_kernel<<<(tot + 255) / 256, 256, 0, stream>>>(h_enc, zr, hc, HID_, tot);
    }
  }

  // ----- memory attention -----
  gemm(h_enc, Wq_b, out + OFF_QUERY, nullptr, B_ * N_, MEMD_, HID_, HID_, MEMD_, MEMD_, 0, 0, 1, 0, 0, 0, 0);
  attention_kernel<<<(B_ * N_) / 256, 256, 0, stream>>>(h_enc, Mem, FCE, out + OFF_QUERY,
                                                        out + OFF_PROTO, out + OFF_POS,
                                                        out + OFF_NEG, WE2, h_cat);

  // ----- decoder supports from W_E (batch-summed gram, D = B*EMB = 256) -----
  gram_softmax_kernel<<<N_, 256, 0, stream>>>(WE2, B_ * EMB_, B_ * EMB_, A_dec, N_);
  gemm(A_dec, A_dec, T2_dec, nullptr, N_, N_, N_, N_, N_, N_, 1, 1, 1, 1, 0, 0, 0);

  hipMemsetAsync(go, 0, BN * sizeof(float), stream);

  // ----- decoder scan -----
  {
    const int c = 2 + DECD_, Bc = B_ * c, Kd = KCH_ * c;  // 98, 3136, 294
    __bf16* Xg0 = Xg;
    __bf16* Xg1 = Xg0 + (size_t)N_ * Bc;
    __bf16* Xg2 = Xg1 + (size_t)N_ * Bc;
    const int packTotal = N_ * Bc;
    const int packBlocks = (packTotal + 255) / 256;
    for (int t = 0; t < HOR_; ++t) {
      pack_kernel<<<packBlocks, 256, 0, stream>>>(Xg0, hist, t, 1, go, h_cat, nullptr, 0, DECD_, c, Bc, packTotal);
      gemm(A_dec,  Xg0, Xg1, nullptr, N_, Bc, N_, N_, Bc, Bc, 0, 1, 1, 1, 0, 0, 0);
      gemm(T2_dec, Xg0, Xg2, nullptr, N_, Bc, N_, N_, Bc, Bc, 0, 1, 1, 1, 0, 0, 0);
      gemm(Xg, dgw_b, zr, dgb, B_ * N_, 2 * DECD_, Kd, 0, 2 * DECD_, 2 * DECD_, 2, 1, 1, 0, 1, c, Bc);
      pack_kernel<<<packBlocks, 256, 0, stream>>>(Xg0, hist, t, 1, go, h_cat, zr, 2 * DECD_, DECD_, c, Bc, packTotal);
      gemm(A_dec,  Xg0, Xg1, nullptr, N_, Bc, N_, N_, Bc, Bc, 0, 1, 1, 1, 0, 0, 0);
      gemm(T2_dec, Xg0, Xg2, nullptr, N_, Bc, N_, N_, Bc, Bc, 0, 1, 1, 1, 0, 0, 0);
      gemm(Xg, duw_b, hc, dub, B_ * N_, DECD_, Kd, 0, DECD_, DECD_, 3, 1, 1, 0, 1, c, Bc);
      int tot = (int)(BN * DECD_);
      gru_combine_kernel<<<(tot + 255) / 256, 256, 0, stream>>>(h_cat, zr, hc, DECD_, tot);
      proj_kernel<<<(B_ * N_ + 255) / 256, 256, 0, stream>>>(h_cat, pw, pb, go, out + OFF_OUT, t);
    }
  }
}
